// BiLSTMCRF_13348758356189
// MI455X (gfx1250) — matrix operand order repaired and checked by reading, not compiled, not run on hardware
//
#include <hip/hip_runtime.h>
#include <hip/hip_bf16.h>
#include <stdint.h>

// ---------------------------------------------------------------------------
// BiLSTM-CRF loss for MI455X (gfx1250, wave32, WMMA)
// B=128, L=512, E=512, H=256, K=64
// ---------------------------------------------------------------------------

#define BB   128          // batch
#define LL   512          // sequence length
#define EE   512          // embedding dim
#define HH   256          // hidden dim
#define KK   64           // num tags
#define G4   (4 * HH)     // 1024 gate width
#define H2   (2 * HH)     // 512 bi-hidden

typedef __attribute__((ext_vector_type(16))) _Float16 v16h;
typedef __attribute__((ext_vector_type(8)))  _Float16 v8h;
typedef __attribute__((ext_vector_type(8)))  float    v8f;

// ---------------------------------------------------------------------------
// WMMA fragment loaders (wave32, 16x16x32 f16 layouts per CDNA5 ISA 7.12.2)
// ---------------------------------------------------------------------------

// A-matrix 16x32 (MxK), row-major source with leading dim `ld` (elements).
// lanes 0-15 : row=lane,  K = {0..7, 16..23}
// lanes 16-31: row=lane-16, K = {8..15, 24..31}
__device__ inline v16h load_a16(const _Float16* __restrict__ base, int ld) {
    const int lane = threadIdx.x & 31;
    const int row  = lane & 15;
    const int koff = (lane & 16) ? 8 : 0;
    const _Float16* p = base + (size_t)row * ld + koff;
    v8h lo = *(const v8h*)(p);        // K = koff .. koff+7
    v8h hi = *(const v8h*)(p + 16);   // K = koff+16 .. koff+23
    v16h r;
#pragma unroll
    for (int i = 0; i < 8; ++i) { r[i] = lo[i]; r[i + 8] = hi[i]; }
    return r;
}

// B-matrix 32x16 (KxN) where memory holds W row-major as (N x K), so
// B[k][n] = W[n][k] and each lane's fragment is two contiguous 8-half runs, 16 apart.
// lanes 0-15 : col=lane,  K = {0..7, 16..23} ; lanes 16-31: col=lane-16, K = {8..15, 24..31}
__device__ inline v16h load_b16(const _Float16* __restrict__ base, int ld) {
    const int lane = threadIdx.x & 31;
    const int col  = lane & 15;
    const int kb   = (lane & 16) ? 8 : 0;
    const _Float16* p = base + (size_t)col * ld + kb;
    v8h lo = *(const v8h*)(p);
    v8h hi = *(const v8h*)(p + 16);
    v16h r;
#pragma unroll
    for (int i = 0; i < 8; ++i) { r[i] = lo[i]; r[i + 8] = hi[i]; }
    return r;
}

__device__ inline v8f wmma_f16(v16h a, v16h b, v8f c) {
    return __builtin_amdgcn_wmma_f32_16x16x32_f16(
        false, a, false, b, (short)0, c, false, false);
}

__device__ inline float sigm(float x) { return 1.0f / (1.0f + __expf(-x)); }

// ---------------------------------------------------------------------------
// Prep kernels
// ---------------------------------------------------------------------------

// x: [B][L][E] f32  ->  xT: [L][B][E] f16
__global__ void cvt_x_kernel(const float* __restrict__ x, _Float16* __restrict__ xT) {
    const long long N = (long long)BB * LL * EE;
    for (long long i = (long long)blockIdx.x * blockDim.x + threadIdx.x; i < N;
         i += (long long)gridDim.x * blockDim.x) {
        int e = (int)(i % EE);
        long long rest = i / EE;
        int l = (int)(rest % LL);
        int b = (int)(rest / LL);
        xT[(size_t)l * BB * EE + (size_t)b * EE + e] = (_Float16)x[i];
    }
}

__global__ void cvt_f16_kernel(const float* __restrict__ src, _Float16* __restrict__ dst, int n) {
    for (int i = blockIdx.x * blockDim.x + threadIdx.x; i < n;
         i += gridDim.x * blockDim.x)
        dst[i] = (_Float16)src[i];
}

__global__ void zero_kernel(unsigned int* __restrict__ p, int nwords) {
    for (int i = blockIdx.x * blockDim.x + threadIdx.x; i < nwords;
         i += gridDim.x * blockDim.x)
        p[i] = 0u;
}

// bias[dir][g] = b_ih + b_hh  (2 x 4H)
__global__ void bias_kernel(const float* __restrict__ bihf, const float* __restrict__ bhhf,
                            const float* __restrict__ bihb, const float* __restrict__ bhhb,
                            float* __restrict__ bias) {
    int i = blockIdx.x * blockDim.x + threadIdx.x;
    if (i < G4)            bias[i] = bihf[i] + bhhf[i];
    else if (i < 2 * G4)   bias[i] = bihb[i - G4] + bhhb[i - G4];
}

// ---------------------------------------------------------------------------
// Fused per-timestep LSTM cell (both directions).
// Wave layout: 256 waves = dir(2) x mTile(B/16=8) x jTile(H/16=16).
// Each wave computes a 16x16 tile of all FOUR gates (i,f,g,o) with WMMA,
// then applies the cell nonlinearity in-register.
// ---------------------------------------------------------------------------
__global__ __launch_bounds__(128) void lstm_step_kernel(
    const _Float16* __restrict__ xT,     // [L][B][E]
    const _Float16* __restrict__ wih,    // [2][4H][E]
    const _Float16* __restrict__ whh,    // [2][4H][H]
    const float*    __restrict__ bias,   // [2][4H]
    _Float16*       __restrict__ hping,  // [2 buf][2 dir][B][H]
    float*          __restrict__ cstate, // [2 dir][B][H]
    _Float16*       __restrict__ hall,   // [L][B][2H]
    int t) {
    const int wg  = blockIdx.x * (blockDim.x >> 5) + (threadIdx.x >> 5);
    const int dir = wg >> 7;
    const int wid = wg & 127;
    const int b0  = (wid >> 4) * 16;     // batch tile origin
    const int j0  = (wid & 15) * 16;     // hidden-column tile origin
    const int tt  = dir ? (LL - 1 - t) : t;

    const _Float16* xrow  = xT + (size_t)tt * BB * EE + (size_t)b0 * EE;
    const _Float16* hprev = hping + (size_t)(t & 1) * 2 * BB * HH
                                  + (size_t)dir * BB * HH + (size_t)b0 * HH;
    const _Float16* wihd  = wih + (size_t)dir * G4 * EE;
    const _Float16* whhd  = whh + (size_t)dir * G4 * HH;

    v8f zero = {};
    v8f acc[4] = {zero, zero, zero, zero};   // gates i, f, g, o

    // input contribution: K over E (x_t is re-read per step; resident in L2)
    for (int k0 = 0; k0 < EE; k0 += 32) {
        v16h a = load_a16(xrow + k0, EE);
#pragma unroll
        for (int q = 0; q < 4; ++q) {
            v16h bm = load_b16(wihd + (size_t)(q * HH + j0) * EE + k0, EE);
            acc[q] = wmma_f16(a, bm, acc[q]);
        }
    }
    // recurrent contribution: K over H
    for (int k0 = 0; k0 < HH; k0 += 32) {
        v16h a = load_a16(hprev + k0, HH);
#pragma unroll
        for (int q = 0; q < 4; ++q) {
            v16h bm = load_b16(whhd + (size_t)(q * HH + j0) * HH + k0, HH);
            acc[q] = wmma_f16(a, bm, acc[q]);
        }
    }

    // C/D layout: VGPR i -> row = i + 8*(lane>=16), col = lane&15
    const int lane = threadIdx.x & 31;
    const int col  = lane & 15;
    const int rb   = (lane & 16) ? 8 : 0;
    const int j    = j0 + col;
    const float bi = bias[dir * G4 + 0 * HH + j];
    const float bf = bias[dir * G4 + 1 * HH + j];
    const float bg = bias[dir * G4 + 2 * HH + j];
    const float bo = bias[dir * G4 + 3 * HH + j];
    float*     cptr  = cstate + (size_t)dir * BB * HH;
    _Float16*  hnext = hping + (size_t)((t + 1) & 1) * 2 * BB * HH + (size_t)dir * BB * HH;
    _Float16*  hout  = hall + (size_t)tt * BB * H2 + (size_t)dir * HH;

#pragma unroll
    for (int i = 0; i < 8; ++i) {
        const int b = b0 + rb + i;
        const float gi = acc[0][i] + bi;
        const float gf = acc[1][i] + bf;
        const float gg = acc[2][i] + bg;
        const float go = acc[3][i] + bo;
        const size_t ci = (size_t)b * HH + j;
        const float c  = cptr[ci];
        const float cn = sigm(gf) * c + sigm(gi) * tanhf(gg);
        const float hn = sigm(go) * tanhf(cn);
        cptr[ci]  = cn;
        hnext[ci] = (_Float16)hn;
        hout[(size_t)b * H2 + j] = (_Float16)hn;
    }
}

// ---------------------------------------------------------------------------
// Emission GEMM: emit[l*B+b][k] = h_all[row] . w_emit[k] + b_emit[k]
// waves = (L*B/16) mTiles x (K/16=4) nTiles
// ---------------------------------------------------------------------------
__global__ __launch_bounds__(128) void emit_kernel(
    const _Float16* __restrict__ hall,   // [L*B][2H]
    const _Float16* __restrict__ wem,    // [K][2H]
    const float*    __restrict__ bem,    // [K]
    float*          __restrict__ emit) { // [L*B][K]
    const int wg = blockIdx.x * (blockDim.x >> 5) + (threadIdx.x >> 5);
    const int r0 = (wg >> 2) * 16;
    const int n0 = (wg & 3) * 16;
    v8f acc = {};
    for (int k0 = 0; k0 < H2; k0 += 32) {
        v16h a  = load_a16(hall + (size_t)r0 * H2 + k0, H2);
        v16h bm = load_b16(wem + (size_t)n0 * H2 + k0, H2);
        acc = wmma_f16(a, bm, acc);
    }
    const int lane = threadIdx.x & 31;
    const int col  = lane & 15;
    const int rb   = (lane & 16) ? 8 : 0;
    const int n    = n0 + col;
    const float bb = bem[n];
#pragma unroll
    for (int i = 0; i < 8; ++i)
        emit[(size_t)(r0 + rb + i) * KK + n] = acc[i] + bb;
}

// ---------------------------------------------------------------------------
// CRF forward + gold score; one workgroup (64 threads) per batch element.
// Transition matrix staged in LDS; fp32 logsumexp.
// emit layout: [L][B][K]
// ---------------------------------------------------------------------------
__global__ __launch_bounds__(64) void crf_kernel(
    const float* __restrict__ emit, const int* __restrict__ labels,
    const float* __restrict__ trans, float* __restrict__ out) {
    __shared__ float Ts[KK * KK];
    __shared__ float dv[KK];
    __shared__ float red[KK];
    const int b = blockIdx.x;
    const int k = threadIdx.x;            // 0..63 (this tag column)

    for (int i = k; i < KK * KK; i += KK) Ts[i] = trans[i];
    dv[k] = emit[(size_t)b * KK + k];     // t = 0
    __syncthreads();

    for (int t = 1; t < LL; ++t) {
        float m = -3.4e38f;
#pragma unroll 8
        for (int p = 0; p < KK; ++p)
            m = fmaxf(m, dv[p] + Ts[p * KK + k]);
        float s = 0.0f;
#pragma unroll 8
        for (int p = 0; p < KK; ++p)
            s += __expf(dv[p] + Ts[p * KK + k] - m);
        const float nd = m + __logf(s) + emit[(size_t)t * BB * KK + (size_t)b * KK + k];
        __syncthreads();
        dv[k] = nd;
        __syncthreads();
    }

    // gold score partials (strided over t)
    float gold = 0.0f;
    for (int t = k; t < LL; t += KK) {
        const int lab = labels[b * LL + t];
        gold += emit[(size_t)t * BB * KK + (size_t)b * KK + lab];
        if (t + 1 < LL)
            gold += trans[lab * KK + labels[b * LL + t + 1]];
    }
    red[k] = gold;
    __syncthreads();

    if (k == 0) {
        float m = dv[0];
        for (int p = 1; p < KK; ++p) m = fmaxf(m, dv[p]);
        float s = 0.0f;
        for (int p = 0; p < KK; ++p) s += __expf(dv[p] - m);
        const float logz = m + __logf(s);
        float g = 0.0f;
        for (int p = 0; p < KK; ++p) g += red[p];
        out[b] = -(g - logz);
    }
}

// ---------------------------------------------------------------------------
// Host launch
// ---------------------------------------------------------------------------
extern "C" void kernel_launch(void* const* d_in, const int* in_sizes, int n_in,
                              void* d_out, int out_size, void* d_ws, size_t ws_size,
                              hipStream_t stream) {
    const float* x       = (const float*)d_in[0];   // (B,L,E)
    const int*   labels  = (const int*)  d_in[1];   // (B,L)
    const float* w_ih_f  = (const float*)d_in[2];   // (4H,E)
    const float* w_hh_f  = (const float*)d_in[3];   // (4H,H)
    const float* b_ih_f  = (const float*)d_in[4];
    const float* b_hh_f  = (const float*)d_in[5];
    const float* w_ih_b  = (const float*)d_in[6];
    const float* w_hh_b  = (const float*)d_in[7];
    const float* b_ih_b  = (const float*)d_in[8];
    const float* b_hh_b  = (const float*)d_in[9];
    const float* w_emit  = (const float*)d_in[10];  // (K,2H)
    const float* b_emit  = (const float*)d_in[11];  // (K)
    const float* trans   = (const float*)d_in[12];  // (K,K)
    float* out = (float*)d_out;                     // (B,)

    // ---- workspace layout (bytes) ----
    uint8_t* ws = (uint8_t*)d_ws;
    size_t off = 0;
    auto take = [&](size_t bytes) {
        uint8_t* p = ws + off;
        off = (off + bytes + 255) & ~(size_t)255;
        return p;
    };
    _Float16* xT     = (_Float16*)take((size_t)LL * BB * EE * 2);   // 67 MB
    _Float16* wih    = (_Float16*)take((size_t)2 * G4 * EE * 2);    //  2 MB
    _Float16* whh    = (_Float16*)take((size_t)2 * G4 * HH * 2);    //  1 MB
    _Float16* wem    = (_Float16*)take((size_t)KK * H2 * 2);        // 64 KB
    float*    bias   = (float*)   take((size_t)2 * G4 * 4);
    _Float16* hping  = (_Float16*)take((size_t)2 * 2 * BB * HH * 2);
    float*    cstate = (float*)   take((size_t)2 * BB * HH * 4);
    _Float16* hall   = (_Float16*)take((size_t)LL * BB * H2 * 2);   // 67 MB
    float*    emit   = (float*)   take((size_t)LL * BB * KK * 4);   // 17 MB
    (void)ws_size; (void)in_sizes; (void)n_in; (void)out_size;

    // ---- prep: f16 conversions + transposition + bias fuse + state zero ----
    cvt_x_kernel<<<8192, 256, 0, stream>>>(x, xT);
    cvt_f16_kernel<<<1024, 256, 0, stream>>>(w_ih_f, wih,            G4 * EE);
    cvt_f16_kernel<<<1024, 256, 0, stream>>>(w_ih_b, wih + G4 * EE,  G4 * EE);
    cvt_f16_kernel<<< 512, 256, 0, stream>>>(w_hh_f, whh,            G4 * HH);
    cvt_f16_kernel<<< 512, 256, 0, stream>>>(w_hh_b, whh + G4 * HH,  G4 * HH);
    cvt_f16_kernel<<<  64, 256, 0, stream>>>(w_emit, wem,            KK * H2);
    bias_kernel<<<8, 256, 0, stream>>>(b_ih_f, b_hh_f, b_ih_b, b_hh_b, bias);
    zero_kernel<<<64, 256, 0, stream>>>((unsigned int*)hping,  (2 * BB * HH * 2) / 4); // buf 0
    zero_kernel<<<64, 256, 0, stream>>>((unsigned int*)cstate, (2 * BB * HH * 4) / 4);

    // ---- sequential BiLSTM scan: one fused kernel per timestep ----
    for (int t = 0; t < LL; ++t)
        lstm_step_kernel<<<64, 128, 0, stream>>>(xT, wih, whh, bias, hping,
                                                 cstate, hall, t);

    // ---- emission GEMM + CRF ----
    emit_kernel<<<4096, 128, 0, stream>>>(hall, wem, b_emit, emit);
    crf_kernel<<<BB, KK, 0, stream>>>(emit, labels, trans, out);
}
